// MFSNetLayer_49761491091624
// MI455X (gfx1250) — matrix operand order repaired and checked by reading, not compiled, not run on hardware
//
#include <hip/hip_runtime.h>
#include <hip/hip_fp16.h>

#define N_TX 32
#define N_SYM 8
#define N_SAMP 256
#define M_DIM 64
#define C_SIG 1.702f
#define LOG2E 1.44269504088896340736f
#define LN2   0.69314718055994530942f

typedef __attribute__((ext_vector_type(16))) _Float16 v16h;
typedef __attribute__((ext_vector_type(8)))  float    v8f;

union FragH { v16h v; unsigned int u[8]; };
union FragF { v8f  v; float        f[8]; };

// counter-based hash RNG -> uniform [0,1)
__device__ __forceinline__ float rng_uniform(unsigned n, unsigned step, unsigned s, unsigned t) {
    unsigned long long x =
        (((unsigned long long)(n * 0x9E3779B9u + step * 0x85EBCA6Bu + 0xC2B2AE35u)) << 32)
        ^ (unsigned long long)(s * 0x27D4EB2Fu + t * 0x165667B1u + 0x1B873593u);
    x ^= x >> 33; x *= 0xff51afd7ed558ccdULL;
    x ^= x >> 33; x *= 0xc4ceb9fe1a85ec53ULL;
    x ^= x >> 33;
    return (float)((x >> 40) & 0xFFFFFFull) * (1.0f / 16777216.0f);
}

__device__ __forceinline__ float sym_val(int k) { return (float)(2 * k - 7); } // {-7,-5,...,7}

// base-2 log-sigmoid: returns logsig(x)/ln2 where x2 = x*log2e is the pre-scaled arg.
// Uses raw hardware TRANS ops: v_exp_f32 (exp2) with neg+abs modifiers, v_log_f32 (log2).
__device__ __forceinline__ float logsig2(float x2) {
    float e = __builtin_amdgcn_exp2f(-fabsf(x2));
    return fminf(x2, 0.f) - __builtin_amdgcn_logf(1.0f + e);
}

__global__ __launch_bounds__(256)
void mfs_kernel(const float* __restrict__ log_qi,   // (N,32,8)
                const float* __restrict__ G,        // (N,64,32)
                const float* __restrict__ sqrt_2rho,// (N,)
                const float* __restrict__ alpha_p,  // scalar
                float* __restrict__ out)            // (N,32,8)
{
    const int n    = blockIdx.x;
    const int tid  = threadIdx.x;
    const int lane = tid & 31;
    const int wave = tid >> 5;

    __shared__ __half Ah[M_DIM * 40];     // C_SIG*log2e*sqrt(2rho)*G in f16, [m][k], stride 40
    __shared__ float  sG32[M_DIM * 33];   // same, f32 copy for alphabet offsets, stride 33
    __shared__ __half Bs[N_SAMP * 40];    // samples transposed: [s][t], stride 40 halves
    __shared__ float  lq[N_TX * N_SYM];   // current log marginals
    __shared__ float  cdf[N_TX * N_SYM];  // per-antenna sampling CDF
    __shared__ float  red[N_SYM];         // cross-wave reduction of logsig sums (base-2 units)

    const float alpha = alpha_p[0];
    // bake sigmoid constant AND log2(e) into the channel: term only appears as
    // logsig(C_SIG * term), and logsig is evaluated in base-2 units.
    const float sq = sqrt_2rho[n] * (C_SIG * LOG2E);

    for (int i = tid; i < M_DIM * N_TX; i += 256) {
        int m = i >> 5, k = i & 31;
        float v = G[(size_t)n * (M_DIM * N_TX) + i] * sq;
        sG32[m * 33 + k] = v;
        Ah[m * 40 + k]   = __float2half(v);
    }
    for (int i = tid; i < N_TX * N_SYM; i += 256) {
        lq[i] = log_qi[(size_t)n * (N_TX * N_SYM) + i];
    }
    __syncthreads();

    // --- this wave's A fragment (constant across all 32 steps) ---
    // wave w: row tile (w & 3), column tiles [(w>>2)*8 .. (w>>2)*8+7]
    const int rtile = wave & 3;
    const int cbase = (wave >> 2) * 8;
    const int m0    = rtile * 16;
    FragH afrag;
    {
        const int am  = m0 + (lane & 15);
        const int ahi = lane >> 4;
        #pragma unroll
        for (int v = 0; v < 8; v++) {
            int kk = (v >> 2) * 16 + ahi * 8 + (v & 3) * 2;  // CDNA5 16-bit A 16x32 layout
            afrag.u[v] = *(const unsigned int*)&Ah[am * 40 + kk];
        }
    }

    // ================= sequential scan over antennas =================
    for (int xi = 0; xi < N_TX; xi++) {
        // --- (1) per-antenna sampling CDFs from current lq ---
        if (tid < N_TX) {
            int t = tid;
            float mx = lq[t * 8];
            #pragma unroll
            for (int k = 1; k < N_SYM; k++) mx = fmaxf(mx, lq[t * 8 + k]);
            float e[N_SYM]; float sum = 0.f;
            #pragma unroll
            for (int k = 0; k < N_SYM; k++) { e[k] = __expf(lq[t * 8 + k] - mx); sum += e[k]; }
            float inv = 1.0f / sum, c = 0.f;
            #pragma unroll
            for (int k = 0; k < N_SYM; k++) { c += e[k] * inv; cdf[t * 8 + k] = c; }
        }
        if (tid < N_SYM) red[tid] = 0.f;
        __syncthreads();

        // --- (2) Monte-Carlo draws: thread = sample index s; B row xi zeroed ---
        {
            const int s = tid;
            #pragma unroll 4
            for (int t = 0; t < N_TX; t++) {
                __half val = __float2half(0.0f);
                if (t != xi) {
                    float u = rng_uniform((unsigned)n, (unsigned)xi, (unsigned)s, (unsigned)t);
                    int k = 0;
                    #pragma unroll
                    for (int kk = 0; kk < N_SYM - 1; kk++) k += (u >= cdf[t * 8 + k]) ? 1 : 0;
                    val = __float2half(sym_val(k));
                }
                Bs[s * 40 + t] = val;
            }
        }
        __syncthreads();

        // --- (3) base = sG(64x32) x samples(32x256) via WMMA f16 (K=32 exact),
        //         then base-2 log-sigmoid with 8 alphabet offsets ---
        float acc[N_SYM];
        #pragma unroll
        for (int j = 0; j < N_SYM; j++) acc[j] = 0.f;

        const int mrow = m0 + (lane >> 4) * 8;  // C/D layout: VGPR p -> M = p + 8*(lane>>4)

        #pragma unroll 2
        for (int ct = 0; ct < 8; ct++) {
            const int c0 = (cbase + ct) * 16;
            FragH bfrag;
            {
                const int bn    = c0 + (lane & 15);
                const int kbase = (lane >> 4) * 8;  // CDNA5 16-bit B 32x16 layout
                #pragma unroll
                for (int v = 0; v < 8; v++)
                    bfrag.u[v] = *(const unsigned int*)&Bs[bn * 40 + kbase + (v >> 2) * 16 + (v & 3) * 2];
            }
            FragF cf;
            #pragma unroll
            for (int p = 0; p < 8; p++) cf.f[p] = 0.f;
            cf.v = __builtin_amdgcn_wmma_f32_16x16x32_f16(
                       false, afrag.v, false, bfrag.v, (short)0, cf.v, false, false);

            #pragma unroll
            for (int p = 0; p < 8; p++) {
                const float g = sG32[(mrow + p) * 33 + xi];  // already scaled by C_SIG*log2e
                const float b = cf.f[p];
                #pragma unroll
                for (int j = 0; j < N_SYM; j++) {
                    acc[j] += logsig2(b + g * sym_val(j));   // ~6 VALU ops, 2 TRANS
                }
            }
        }
        #pragma unroll
        for (int j = 0; j < N_SYM; j++) atomicAdd(&red[j], acc[j]);
        __syncthreads();

        // --- (4) damped update of row xi: ln2 * mean over S of sum over m ---
        if (tid < N_SYM) {
            float ex = red[tid] * (LN2 / (float)N_SAMP);
            lq[xi * 8 + tid] = (1.0f - alpha) * lq[xi * 8 + tid] + alpha * ex;
        }
        __syncthreads();

        // --- (5) subtract per-row max (all rows), as in reference ---
        if (tid < N_TX) {
            float mx = lq[tid * 8];
            #pragma unroll
            for (int k = 1; k < N_SYM; k++) mx = fmaxf(mx, lq[tid * 8 + k]);
            #pragma unroll
            for (int k = 0; k < N_SYM; k++) lq[tid * 8 + k] -= mx;
        }
        __syncthreads();
    }

    for (int i = tid; i < N_TX * N_SYM; i += 256) {
        out[(size_t)n * (N_TX * N_SYM) + i] = lq[i];
    }
}

extern "C" void kernel_launch(void* const* d_in, const int* in_sizes, int n_in,
                              void* d_out, int out_size, void* d_ws, size_t ws_size,
                              hipStream_t stream) {
    const float* log_qi    = (const float*)d_in[0];
    const float* G         = (const float*)d_in[1];
    const float* sqrt_2rho = (const float*)d_in[2];
    const float* alpha     = (const float*)d_in[3];
    float* out = (float*)d_out;

    const int N = in_sizes[0] / (N_TX * N_SYM);   // 512
    mfs_kernel<<<N, 256, 0, stream>>>(log_qi, G, sqrt_2rho, alpha, out);
    (void)d_ws; (void)ws_size; (void)out_size; (void)n_in;
}
